// RotarySelfAttention_36670430773964
// MI455X (gfx1250) — matrix operand order repaired and checked by reading, not compiled, not run on hardware
//
#include <hip/hip_runtime.h>
#include <hip/hip_bf16.h>

// ---------------------------------------------------------------------------
// Problem constants (from reference)
// ---------------------------------------------------------------------------
#define T_SEQ 4096
#define C_DIM 1024
#define NHQ   16
#define NKV   4
#define HDIM  64
#define WIN   1024

typedef __attribute__((ext_vector_type(16))) __bf16 v16bf;
typedef __attribute__((ext_vector_type(8)))  __bf16 v8bf;
typedef __attribute__((ext_vector_type(8)))  float  v8f;

#if __has_builtin(__builtin_amdgcn_global_load_async_to_lds_b128) && \
    __has_builtin(__builtin_amdgcn_s_wait_asynccnt)
#define USE_ASYNC_LDS 1
typedef int v4i __attribute__((ext_vector_type(4)));
typedef __attribute__((address_space(1))) v4i glob_v4i;   // global int4
typedef __attribute__((address_space(3))) v4i lds_v4i;    // LDS int4

__device__ __forceinline__ glob_v4i* to_global_v4i(const void* p) {
    return (glob_v4i*)(unsigned long long)p;
}
__device__ __forceinline__ lds_v4i* to_lds_v4i(void* p) {
    return (lds_v4i*)(unsigned int)(unsigned long long)p;
}
#else
#define USE_ASYNC_LDS 0
#endif

__device__ __forceinline__ v16bf join8(v8bf lo, v8bf hi) {
    v16bf r;
#pragma unroll
    for (int i = 0; i < 8; ++i) { r[i] = lo[i]; r[i + 8] = hi[i]; }
    return r;
}

// ---------------------------------------------------------------------------
// 1) f32 -> bf16 convert
// ---------------------------------------------------------------------------
__global__ __launch_bounds__(256) void cvt_f32_bf16(const float* __restrict__ s,
                                                    __bf16* __restrict__ d, int n) {
    int i = blockIdx.x * 256 + threadIdx.x;
    if (i < n) d[i] = (__bf16)s[i];
}

// ---------------------------------------------------------------------------
// 2) GEMM: Out[M,N] (f32) = X[M,K](bf16) * W[N,K]^T(bf16)
//    Workgroup tile 128x64, 8 waves, wave tile 32x32 (2x2 WMMA 16x16x32 acc).
//    Tiles staged into padded LDS (40 elems = 80B rows -> 16B-aligned b128)
//    via CDNA5 async global->LDS copies when available (ASYNCcnt-tracked).
// ---------------------------------------------------------------------------
#define GBM 128
#define GBN 64
#define GKS 32
#define LDSP 40

__global__ __launch_bounds__(256)
void gemm_bf16_f32(const __bf16* __restrict__ X, const __bf16* __restrict__ W,
                   float* __restrict__ Out, int M, int N, int K) {
    __shared__ __align__(16) __bf16 Xs[GBM * LDSP];
    __shared__ __align__(16) __bf16 Ws[GBN * LDSP];

    const int tid  = threadIdx.x;
    const int lane = tid & 31;
    const int wid  = tid >> 5;
    const int hi   = lane >> 4;     // 0/1 half of wave
    const int l15  = lane & 15;
    const int wm   = wid & 3;       // 4 M-groups of 32
    const int wn   = wid >> 2;      // 2 N-groups of 32
    const int mBase = blockIdx.y * GBM;
    const int nBase = blockIdx.x * GBN;

    // Per-thread staging chunk coordinates (8 bf16 = 16B per chunk).
    const int xrow0 = tid >> 2,            xcol0 = (tid & 3) * 8;
    const int xrow1 = (tid + 256) >> 2,    xcol1 = xcol0;
    const int wrow  = tid >> 2,            wcol  = (tid & 3) * 8;

    v8f acc[2][2] = {};

    for (int kt = 0; kt < K; kt += GKS) {
        __syncthreads();   // previous iteration's fragment reads complete
#if USE_ASYNC_LDS
        // CDNA5 direct global->LDS async copies (no VGPR staging).
        __builtin_amdgcn_global_load_async_to_lds_b128(
            to_global_v4i(X + (size_t)(mBase + xrow0) * K + kt + xcol0),
            to_lds_v4i(Xs + xrow0 * LDSP + xcol0), 0, 0);
        __builtin_amdgcn_global_load_async_to_lds_b128(
            to_global_v4i(X + (size_t)(mBase + xrow1) * K + kt + xcol1),
            to_lds_v4i(Xs + xrow1 * LDSP + xcol1), 0, 0);
        __builtin_amdgcn_global_load_async_to_lds_b128(
            to_global_v4i(W + (size_t)(nBase + wrow) * K + kt + wcol),
            to_lds_v4i(Ws + wrow * LDSP + wcol), 0, 0);
        __builtin_amdgcn_s_wait_asynccnt(0);
#else
        {
            v8bf vx0 = *(const v8bf*)(X + (size_t)(mBase + xrow0) * K + kt + xcol0);
            v8bf vx1 = *(const v8bf*)(X + (size_t)(mBase + xrow1) * K + kt + xcol1);
            v8bf vw  = *(const v8bf*)(W + (size_t)(nBase + wrow) * K + kt + wcol);
            *(v8bf*)(Xs + xrow0 * LDSP + xcol0) = vx0;
            *(v8bf*)(Xs + xrow1 * LDSP + xcol1) = vx1;
            *(v8bf*)(Ws + wrow * LDSP + wcol)   = vw;
        }
#endif
        __syncthreads();   // all waves' LDS writes visible

        // A fragments: row m = l15, elems e -> k = (e>>3)*16 + hi*8 + (e&7)
        v16bf a[2], b[2];
#pragma unroll
        for (int mt = 0; mt < 2; ++mt) {
            const __bf16* base = Xs + (wm * 32 + mt * 16 + l15) * LDSP + hi * 8;
            a[mt] = join8(*(const v8bf*)base, *(const v8bf*)(base + 16));
        }
        // B fragments: col n = l15, elems e -> k = (e>>3)*16 + hi*8 + (e&7)
#pragma unroll
        for (int nt = 0; nt < 2; ++nt) {
            const __bf16* base = Ws + (wn * 32 + nt * 16 + l15) * LDSP + hi * 8;
            b[nt] = join8(*(const v8bf*)base, *(const v8bf*)(base + 16));
        }
#pragma unroll
        for (int mt = 0; mt < 2; ++mt)
#pragma unroll
            for (int nt = 0; nt < 2; ++nt)
                acc[mt][nt] = __builtin_amdgcn_wmma_f32_16x16x32_bf16(
                    false, a[mt], false, b[nt], (short)0, acc[mt][nt], false, false);
    }

    // D layout: n = l15, m = hi*8 + v
#pragma unroll
    for (int mt = 0; mt < 2; ++mt)
#pragma unroll
        for (int nt = 0; nt < 2; ++nt)
#pragma unroll
            for (int v = 0; v < 8; ++v) {
                int row = mBase + wm * 32 + mt * 16 + hi * 8 + v;
                int col = nBase + wn * 32 + nt * 16 + l15;
                Out[(size_t)row * N + col] = acc[mt][nt][v];
            }
}

// ---------------------------------------------------------------------------
// 3) RMSNorm + RoPE for Q/K: one wave per (t, head); lane owns rope pair
//    (2*lane, 2*lane+1), freq index = lane (HD/2 == 32 == wave size).
// ---------------------------------------------------------------------------
__global__ __launch_bounds__(256)
void qk_post(const float* __restrict__ src, __bf16* __restrict__ dst,
             const float* __restrict__ nw, const float* __restrict__ fc,
             const float* __restrict__ fs, int nheads) {
    int gw   = (blockIdx.x * 256 + threadIdx.x) >> 5;
    int lane = threadIdx.x & 31;
    int t = gw / nheads, h = gw % nheads;
    const float* p = src + (size_t)t * nheads * HDIM + h * HDIM;
    float vr = p[2 * lane], vi = p[2 * lane + 1];
    float ss = vr * vr + vi * vi;
#pragma unroll
    for (int m = 16; m >= 1; m >>= 1) ss += __shfl_xor(ss, m, 32);
    float r = rsqrtf(ss * (1.0f / HDIM) + 1e-5f);
    vr *= r * nw[2 * lane];
    vi *= r * nw[2 * lane + 1];
    float c = fc[t * 32 + lane], s = fs[t * 32 + lane];
    float orr = vr * c - vi * s;
    float oii = vr * s + vi * c;
    __bf16* q = dst + (size_t)t * nheads * HDIM + h * HDIM + 2 * lane;
    q[0] = (__bf16)orr;
    q[1] = (__bf16)oii;
}

// ---------------------------------------------------------------------------
// 4) V: f32 [t][256] -> bf16 transposed [d][t] so PV B-frags load contiguously
// ---------------------------------------------------------------------------
__global__ __launch_bounds__(256)
void v_post(const float* __restrict__ vf, __bf16* __restrict__ vt) {
    int idx = blockIdx.x * 256 + threadIdx.x;   // T_SEQ * 256 elems
    int t = idx >> 8, d = idx & 255;
    vt[(size_t)d * T_SEQ + t] = (__bf16)vf[idx];
}

// ---------------------------------------------------------------------------
// 5) Sliding-window flash attention. One wave per (head, 16-row q block).
//    S = Q*K^T via 16x16x32 bf16 WMMA (f32 acc), online softmax with
//    16-lane xor-shuffle row reductions, P staged bf16 through a private
//    LDS slice (D-layout -> A-layout), O += P*V via WMMA on transposed V.
// ---------------------------------------------------------------------------
__global__ __launch_bounds__(256)
void attn_win(const __bf16* __restrict__ Q, const __bf16* __restrict__ Kb,
              const __bf16* __restrict__ Vt, __bf16* __restrict__ Y) {
    __shared__ __align__(16) __bf16 plds[8][16 * LDSP];   // 10,240 B

    const int wid  = threadIdx.x >> 5;
    const int lane = threadIdx.x & 31;
    const int task = blockIdx.x * 8 + wid;      // 0..4095
    const int h    = task >> 8;                 // head
    const int qb   = (task & 255) * 16;         // q row base
    const int kv   = h >> 2;
    const int hi   = lane >> 4;
    const int l15  = lane & 15;
    __bf16* P = plds[wid];

    // Q A-fragments for HD=64 (two K=32 steps), kept in registers all loop.
    const __bf16* qrow = Q + (size_t)(qb + l15) * (NHQ * HDIM) + h * HDIM;
    v16bf qa[2];
#pragma unroll
    for (int kc = 0; kc < 2; ++kc) {
        const __bf16* b = qrow + kc * 32 + hi * 8;
        qa[kc] = join8(*(const v8bf*)b, *(const v8bf*)(b + 16));
    }

    v8f O[4] = {};
    float mrow[8], lrow[8];
#pragma unroll
    for (int v = 0; v < 8; ++v) { mrow[v] = -1e30f; lrow[v] = 0.0f; }

    int sLo = qb - (WIN - 16);
    if (sLo < 0) sLo = 0;
    sLo &= ~31;

    for (int s0 = sLo; s0 <= qb + 15; s0 += 32) {
        // Prefetch next key/value block into GL2 (global_prefetch_b8).
        {
            int sp = s0 + 32;
            if (sp > T_SEQ - 32) sp = T_SEQ - 32;
            __builtin_prefetch(Kb + (size_t)(sp + l15) * (NKV * HDIM) + kv * HDIM, 0, 1);
            __builtin_prefetch(Vt + (size_t)(kv * HDIM + l15) * T_SEQ + sp, 0, 1);
        }
        // ---- S = Q * K^T  (two 16-col key tiles) ----
        v8f sc[2];
#pragma unroll
        for (int nt = 0; nt < 2; ++nt) {
            const __bf16* krow =
                Kb + (size_t)(s0 + nt * 16 + l15) * (NKV * HDIM) + kv * HDIM;
            v8f z = {};
#pragma unroll
            for (int kc = 0; kc < 2; ++kc) {
                const __bf16* b = krow + kc * 32 + hi * 8;
                v16bf kb = join8(*(const v8bf*)b, *(const v8bf*)(b + 16));
                z = __builtin_amdgcn_wmma_f32_16x16x32_bf16(
                    false, qa[kc], false, kb, (short)0, z, false, false);
            }
            sc[nt] = z;
        }
        // ---- scale + sliding-window causal mask ----
#pragma unroll
        for (int nt = 0; nt < 2; ++nt)
#pragma unroll
            for (int v = 0; v < 8; ++v) {
                int i = qb + hi * 8 + v;
                int j = s0 + nt * 16 + l15;
                float val = sc[nt][v] * 0.125f;   // 1/sqrt(64)
                bool ok = (i >= j) && (i - j < WIN);
                sc[nt][v] = ok ? val : -1e30f;
            }
        // ---- online softmax: row max / rescale ----
        float mnew[8], scl[8];
#pragma unroll
        for (int v = 0; v < 8; ++v) {
            float r = fmaxf(sc[0][v], sc[1][v]);
            r = fmaxf(r, __shfl_xor(r, 1, 32));
            r = fmaxf(r, __shfl_xor(r, 2, 32));
            r = fmaxf(r, __shfl_xor(r, 4, 32));
            r = fmaxf(r, __shfl_xor(r, 8, 32));
            mnew[v] = fmaxf(mrow[v], r);
            scl[v]  = __expf(mrow[v] - mnew[v]);
            mrow[v] = mnew[v];
        }
        // ---- P = exp(S - m), stage to LDS in A-source order ----
#pragma unroll
        for (int nt = 0; nt < 2; ++nt)
#pragma unroll
            for (int v = 0; v < 8; ++v) {
                float sv = sc[nt][v];
                float p  = (sv > -1e29f) ? __expf(sv - mnew[v]) : 0.0f;
                sc[nt][v] = p;
                P[(hi * 8 + v) * LDSP + nt * 16 + l15] = (__bf16)p;
            }
#pragma unroll
        for (int v = 0; v < 8; ++v) {
            float rs = sc[0][v] + sc[1][v];
            rs += __shfl_xor(rs, 1, 32);
            rs += __shfl_xor(rs, 2, 32);
            rs += __shfl_xor(rs, 4, 32);
            rs += __shfl_xor(rs, 8, 32);
            lrow[v] = lrow[v] * scl[v] + rs;
        }
#pragma unroll
        for (int nt = 0; nt < 4; ++nt)
#pragma unroll
            for (int v = 0; v < 8; ++v) O[nt][v] *= scl[v];

        // ---- O += P * V  (K = 32 keys, 4 hd tiles of 16) ----
        // Same-wave LDS RAW: DS pipe is in-order, compiler emits s_wait_dscnt.
        const __bf16* pb = P + l15 * LDSP + hi * 8;
        v16bf pa = join8(*(const v8bf*)pb, *(const v8bf*)(pb + 16));
#pragma unroll
        for (int nt = 0; nt < 4; ++nt) {
            const __bf16* vrow =
                Vt + (size_t)(kv * HDIM + nt * 16 + l15) * T_SEQ + s0 + hi * 8;
            v16bf vb = join8(*(const v8bf*)vrow, *(const v8bf*)(vrow + 16));
            O[nt] = __builtin_amdgcn_wmma_f32_16x16x32_bf16(
                false, pa, false, vb, (short)0, O[nt], false, false);
        }
    }

    // ---- normalize and store Y (bf16, [T][NH*HD]) ----
    float inv[8];
#pragma unroll
    for (int v = 0; v < 8; ++v) inv[v] = (lrow[v] > 0.0f) ? 1.0f / lrow[v] : 0.0f;
#pragma unroll
    for (int nt = 0; nt < 4; ++nt)
#pragma unroll
        for (int v = 0; v < 8; ++v) {
            int row = qb + hi * 8 + v;
            int col = h * HDIM + nt * 16 + l15;
            Y[(size_t)row * (NHQ * HDIM) + col] = (__bf16)(O[nt][v] * inv[v]);
        }
}

// ---------------------------------------------------------------------------
// Host launcher
// ---------------------------------------------------------------------------
extern "C" void kernel_launch(void* const* d_in, const int* in_sizes, int n_in,
                              void* d_out, int out_size, void* d_ws, size_t ws_size,
                              hipStream_t stream) {
    const float* x   = (const float*)d_in[0];
    const float* wq  = (const float*)d_in[1];
    const float* wk  = (const float*)d_in[2];
    const float* wv  = (const float*)d_in[3];
    const float* wo  = (const float*)d_in[4];
    const float* qnw = (const float*)d_in[5];
    const float* knw = (const float*)d_in[6];
    const float* fc  = (const float*)d_in[7];
    const float* fs  = (const float*)d_in[8];
    float* out = (float*)d_out;

    // Workspace carve-up (256B aligned)
    auto align256 = [](size_t v) { return (v + 255) & ~(size_t)255; };
    char* ws = (char*)d_ws;
    size_t off = 0;
    auto take = [&](size_t bytes) { char* p = ws + off; off += align256(bytes); return p; };

    __bf16* xb  = (__bf16*)take((size_t)T_SEQ * C_DIM * 2);
    __bf16* wqb = (__bf16*)take((size_t)C_DIM * C_DIM * 2);
    __bf16* wkb = (__bf16*)take((size_t)NKV * HDIM * C_DIM * 2);
    __bf16* wvb = (__bf16*)take((size_t)NKV * HDIM * C_DIM * 2);
    __bf16* wob = (__bf16*)take((size_t)C_DIM * C_DIM * 2);
    float*  qf  = (float*)take((size_t)T_SEQ * C_DIM * 4);
    float*  kf  = (float*)take((size_t)T_SEQ * NKV * HDIM * 4);
    float*  vf  = (float*)take((size_t)T_SEQ * NKV * HDIM * 4);
    __bf16* qbf = (__bf16*)take((size_t)T_SEQ * C_DIM * 2);
    __bf16* kbf = (__bf16*)take((size_t)T_SEQ * NKV * HDIM * 2);
    __bf16* vtb = (__bf16*)take((size_t)NKV * HDIM * T_SEQ * 2);
    __bf16* yb  = (__bf16*)take((size_t)T_SEQ * C_DIM * 2);
    (void)ws_size; (void)n_in; (void)in_sizes; (void)out_size;

    // 1) convert inputs to bf16
    {
        int n;
        n = T_SEQ * C_DIM;
        cvt_f32_bf16<<<(n + 255) / 256, 256, 0, stream>>>(x, xb, n);
        n = C_DIM * C_DIM;
        cvt_f32_bf16<<<(n + 255) / 256, 256, 0, stream>>>(wq, wqb, n);
        n = NKV * HDIM * C_DIM;
        cvt_f32_bf16<<<(n + 255) / 256, 256, 0, stream>>>(wk, wkb, n);
        cvt_f32_bf16<<<(n + 255) / 256, 256, 0, stream>>>(wv, wvb, n);
        n = C_DIM * C_DIM;
        cvt_f32_bf16<<<(n + 255) / 256, 256, 0, stream>>>(wo, wob, n);
    }

    // 2) QKV projections (WMMA GEMM)
    gemm_bf16_f32<<<dim3(C_DIM / GBN, T_SEQ / GBM), 256, 0, stream>>>(
        xb, wqb, qf, T_SEQ, C_DIM, C_DIM);
    gemm_bf16_f32<<<dim3(NKV * HDIM / GBN, T_SEQ / GBM), 256, 0, stream>>>(
        xb, wkb, kf, T_SEQ, NKV * HDIM, C_DIM);
    gemm_bf16_f32<<<dim3(NKV * HDIM / GBN, T_SEQ / GBM), 256, 0, stream>>>(
        xb, wvb, vf, T_SEQ, NKV * HDIM, C_DIM);

    // 3) RMSNorm + RoPE -> bf16 (one wave per (t, head))
    qk_post<<<(T_SEQ * NHQ) / 8, 256, 0, stream>>>(qf, qbf, qnw, fc, fs, NHQ);
    qk_post<<<(T_SEQ * NKV) / 8, 256, 0, stream>>>(kf, kbf, knw, fc, fs, NKV);

    // 4) V -> bf16 transposed [d][t]
    v_post<<<(T_SEQ * NKV * HDIM) / 256, 256, 0, stream>>>(vf, vtb);

    // 5) sliding-window flash attention (one wave per head x 16-row block)
    attn_win<<<(NHQ * (T_SEQ / 16)) / 8, 256, 0, stream>>>(qbf, kbf, vtb, yb);

    // 6) output projection -> f32 d_out
    gemm_bf16_f32<<<dim3(C_DIM / GBN, T_SEQ / GBM), 256, 0, stream>>>(
        yb, wob, out, T_SEQ, C_DIM, C_DIM);
}
